// MultiHeadAttention_26396869001639
// MI455X (gfx1250) — matrix operand order repaired and checked by reading, not compiled, not run on hardware
//
#include <hip/hip_runtime.h>
#include <hip/hip_bf16.h>

typedef __attribute__((ext_vector_type(16))) __bf16       v16bf;
typedef __attribute__((ext_vector_type(8)))  float        v8f;
typedef __attribute__((ext_vector_type(4)))  unsigned int u32x4;
typedef __attribute__((ext_vector_type(8)))  int          i32x8;
typedef __attribute__((ext_vector_type(4)))  int          i32x4;
typedef __attribute__((ext_vector_type(4)))  int          v4i;

// ---------------------------------------------------------------------------
// Feature probes (compile-safe on both ROCm 7.2 / amdgpu-toolchain)
// ---------------------------------------------------------------------------
#if __has_builtin(__builtin_amdgcn_global_load_async_to_lds_b128)
#define HAVE_ASYNC_LDS 1
#else
#define HAVE_ASYNC_LDS 0
#endif
#if __has_builtin(__builtin_amdgcn_tensor_load_to_lds)
#define HAVE_TDM 1
#else
#define HAVE_TDM 0
#endif

// 16-byte copy global -> LDS (async when available, ASYNCcnt-tracked)
__device__ __forceinline__ void copy16(void* lds_dst, const void* gsrc) {
#if HAVE_ASYNC_LDS
    __builtin_amdgcn_global_load_async_to_lds_b128(
        (__attribute__((address_space(1))) v4i*)gsrc,
        (__attribute__((address_space(3))) v4i*)lds_dst, 0, 0);
#else
    *(uint4*)lds_dst = *(const uint4*)gsrc;
#endif
}

__device__ __forceinline__ void wait_async_lds() {
#if HAVE_ASYNC_LDS
#if __has_builtin(__builtin_amdgcn_s_wait_asynccnt)
    __builtin_amdgcn_s_wait_asynccnt(0);
#else
    asm volatile("s_wait_asynccnt 0x0" ::: "memory");
#endif
#endif
}

__device__ __forceinline__ void wait_tensor() {
#if HAVE_TDM
#if __has_builtin(__builtin_amdgcn_s_wait_tensorcnt)
    __builtin_amdgcn_s_wait_tensorcnt(0);
#else
    asm volatile("s_wait_tensorcnt 0x0" ::: "memory");
#endif
#endif
}

#if HAVE_TDM
// TDM: load a 64x64 bf16 tile (row stride `stride` elements) into LDS at
// byte offset lds_off.  2D descriptor, groups 2/3 zero (dims 3/4 unused).
__device__ __forceinline__ void tdm_load_64x64_bf16(unsigned lds_off,
                                                    const void* gsrc,
                                                    unsigned stride) {
    unsigned long long ga = (unsigned long long)(size_t)gsrc;
    u32x4 g0 = (u32x4)0u;
    g0[0] = 1u;                                   // count=1, user descriptor
    g0[1] = lds_off;                              // lds_addr
    g0[2] = (unsigned)ga;                         // global_addr[31:0]
    g0[3] = (unsigned)((ga >> 32) & 0x01FFFFFFull) | (2u << 30);  // addr hi | type=2
    i32x8 g1 = (i32x8)0;
    g1[0] = 0x00010000;          // wg_mask=0 | data_size=1 (2 bytes)
    g1[1] = (int)(64u << 16);    // tensor_dim0[15:0] = 64
    g1[2] = (int)(64u << 16);    // tensor_dim0 hi = 0 | tensor_dim1[15:0] = 64
    g1[3] = (int)(64u << 16);    // tensor_dim1 hi = 0 | tile_dim0 = 64
    g1[4] = 64;                  // tile_dim1 = 64 | tile_dim2 = 0
    g1[5] = (int)stride;         // tensor_dim0_stride[31:0]
    i32x4 g2 = (i32x4)0, g3 = (i32x4)0;
#if __clang_major__ >= 23
    i32x8 g4 = (i32x8)0;
    __builtin_amdgcn_tensor_load_to_lds(g0, g1, g2, g3, g4, 0);
#else
    __builtin_amdgcn_tensor_load_to_lds(g0, g1, g2, g3, 0);
#endif
}
#endif

// ---------------------------------------------------------------------------
// Cross-lane XOR reduction via ds_swizzle_b32 (group-of-32 mode, immediate
// pattern: offset = xor_mask<<10 | and_mask(0x1F); no address VGPR needed).
// XOR masks {1,2,4,8} stay within each 16-lane half (rows replicated there).
// ---------------------------------------------------------------------------
template <int XORM>
__device__ __forceinline__ float swz_xor(float v) {
    return __int_as_float(
        __builtin_amdgcn_ds_swizzle(__float_as_int(v), (XORM << 10) | 0x1F));
}
__device__ __forceinline__ float half_reduce_max(float v) {
    v = fmaxf(v, swz_xor<1>(v));
    v = fmaxf(v, swz_xor<2>(v));
    v = fmaxf(v, swz_xor<4>(v));
    v = fmaxf(v, swz_xor<8>(v));
    return v;
}
__device__ __forceinline__ float half_reduce_sum(float v) {
    v += swz_xor<1>(v);
    v += swz_xor<2>(v);
    v += swz_xor<4>(v);
    v += swz_xor<8>(v);
    return v;
}

// ---------------------------------------------------------------------------
// WMMA helper: D = A(16x32 bf16) x B(32x16 bf16) + C(16x16 f32)
// ---------------------------------------------------------------------------
__device__ __forceinline__ v8f wmma_bf16(v16bf a, v16bf b, v8f c) {
    return __builtin_amdgcn_wmma_f32_16x16x32_bf16(
        false, a, false, b, (short)0, c, false, false);
}

// A-fragment (16x32 MxK) from LDS, row stride ld.  ISA wave32 layout:
// element j -> K = (j>>3)*16 + (lane>>4)*8 + (j&7), row = lane&15.
__device__ __forceinline__ v16bf load_frag_a(const __bf16* p, int ld, int lane) {
    const int row = lane & 15, kh = lane >> 4;
    v16bf a;
#pragma unroll
    for (int j = 0; j < 16; ++j) {
        const int k = ((j >> 3) << 4) + (kh << 3) + (j & 7);
        a[j] = p[row * ld + k];
    }
    return a;
}

// B-fragment (32x16 KxN), memory N-major: p[n*ld + k].  element j -> K = (j>>3)*16 + (lane>>4)*8 + (j&7).
__device__ __forceinline__ v16bf load_frag_b_nk(const __bf16* p, int ld, int lane) {
    const int n = lane & 15, kh = lane >> 4;
    v16bf b;
#pragma unroll
    for (int j = 0; j < 16; ++j) b[j] = p[n * ld + ((j >> 3) << 4) + (kh << 3) + (j & 7)];
    return b;
}

// B-fragment (32x16 KxN), memory K-major: p[k*ld + n]  (V tiles: [key][hd]).
__device__ __forceinline__ v16bf load_frag_b_kn(const __bf16* p, int ld, int lane) {
    const int n = lane & 15, kh = lane >> 4;
    v16bf b;
#pragma unroll
    for (int j = 0; j < 16; ++j) b[j] = p[(((j >> 3) << 4) + (kh << 3) + (j & 7)) * ld + n];
    return b;
}

// ---------------------------------------------------------------------------
// f32 -> bf16 cast
// ---------------------------------------------------------------------------
__global__ void cast_f32_bf16(const float* __restrict__ src,
                              __bf16* __restrict__ dst, int n) {
    int i = blockIdx.x * blockDim.x + threadIdx.x;
    const int stride = gridDim.x * blockDim.x;
    for (; i < n; i += stride) dst[i] = (__bf16)src[i];
}

// ---------------------------------------------------------------------------
// GEMM: C[M,N] = A[M,K] @ W[N,K]^T (+bias).  128x128 block, BK=32,
// double-buffered LDS with async global->LDS staging, 8 waves, 8 WMMA/K-step.
// ---------------------------------------------------------------------------
template <bool OUT_BF16>
__global__ void __launch_bounds__(256)
gemm_xwt(const __bf16* __restrict__ A, const __bf16* __restrict__ W,
         void* __restrict__ Cout, const float* __restrict__ bias,
         int M, int N, int K) {
    __shared__ __bf16 As[2][128 * 32];
    __shared__ __bf16 Ws[2][128 * 32];

    const int tid   = threadIdx.x;
    const int lane  = tid & 31;
    const int wid   = tid >> 5;
    const int waveM = wid >> 1;   // 0..3
    const int waveN = wid & 1;    // 0..1
    const int mBase = blockIdx.y * 128;
    const int nBase = blockIdx.x * 128;

    v8f acc[2][4] = {};

    auto stage = [&](int buf, int k0) {
        for (int i = tid; i < 512; i += 256) {
            const int row = i >> 2, seg = i & 3;
            copy16(&As[buf][row * 32 + seg * 8],
                   &A[(size_t)(mBase + row) * K + k0 + seg * 8]);
            copy16(&Ws[buf][row * 32 + seg * 8],
                   &W[(size_t)(nBase + row) * K + k0 + seg * 8]);
        }
    };

    stage(0, 0);
    wait_async_lds();
    __syncthreads();

    int cur = 0;
    for (int k0 = 0; k0 < K; k0 += 32) {
        if (k0 + 32 < K) stage(cur ^ 1, k0 + 32);

        v16bf aF[2], bF[4];
#pragma unroll
        for (int mt = 0; mt < 2; ++mt)
            aF[mt] = load_frag_a(&As[cur][(waveM * 32 + mt * 16) * 32], 32, lane);
#pragma unroll
        for (int nt = 0; nt < 4; ++nt)
            bF[nt] = load_frag_b_nk(&Ws[cur][(waveN * 64 + nt * 16) * 32], 32, lane);
#pragma unroll
        for (int mt = 0; mt < 2; ++mt)
#pragma unroll
            for (int nt = 0; nt < 4; ++nt)
                acc[mt][nt] = wmma_bf16(aF[mt], bF[nt], acc[mt][nt]);

        wait_async_lds();
        __syncthreads();
        cur ^= 1;
    }

    const int kh = lane >> 4, nl = lane & 15;
#pragma unroll
    for (int mt = 0; mt < 2; ++mt)
#pragma unroll
        for (int nt = 0; nt < 4; ++nt) {
            const int n = nBase + waveN * 64 + nt * 16 + nl;
            const float bs = (!OUT_BF16 && bias) ? bias[n] : 0.0f;
#pragma unroll
            for (int r = 0; r < 8; ++r) {
                const int m = mBase + waveM * 32 + mt * 16 + r + 8 * kh;
                const float v = acc[mt][nt][r];
                if (OUT_BF16)
                    ((__bf16*)Cout)[(size_t)m * N + n] = (__bf16)v;
                else
                    ((float*)Cout)[(size_t)m * N + n] = v + bs;
            }
        }
}

// ---------------------------------------------------------------------------
// Causal flash attention per (b,h).  Grid (S/64, B*H), 128 threads = 4 waves.
// Dynamic LDS: Kb[2] @0/8K, Vb[2] @16K/24K, P @32K + wave*2K  (40KB total).
// K/V tiles staged by the Tensor Data Mover (wave0 -> K, wave1 -> V) when
// available, double-buffered against WMMA compute.
// ---------------------------------------------------------------------------
__global__ void __launch_bounds__(128)
flash_attn(const __bf16* __restrict__ Q, const __bf16* __restrict__ Kg,
           const __bf16* __restrict__ Vg, __bf16* __restrict__ ctx) {
    constexpr int S = 2048, D = 1024, HD = 64, H = 16;
    extern __shared__ char smem[];

    const int tid  = threadIdx.x;
    const int lane = tid & 31;
    const int w    = tid >> 5;
    const int kh   = lane >> 4;
    const int nl   = lane & 15;
    const int qb   = blockIdx.x;
    const int bh   = blockIdx.y;
    const int bidx = bh >> 4, h = bh & (H - 1);
    const int qBase = qb * 64;
    const size_t base = ((size_t)bidx * S) * D + (size_t)h * HD;

    auto KbP = [&](int buf) { return (__bf16*)(smem + (unsigned)buf * 8192u); };
    auto VbP = [&](int buf) { return (__bf16*)(smem + 16384u + (unsigned)buf * 8192u); };
    __bf16* Pw = (__bf16*)(smem + 32768u + (unsigned)w * 2048u);

    // Q A-fragments, pre-scaled by 1/sqrt(HD)=0.125 (power of 2 -> exact in bf16)
    const int qrow = qBase + 16 * w + nl;
    v16bf qF[2];
#pragma unroll
    for (int f = 0; f < 2; ++f)
#pragma unroll
        for (int j = 0; j < 16; ++j) {
            const int k = f * 32 + ((j >> 3) << 4) + (kh << 3) + (j & 7);
            qF[f][j] = (__bf16)(0.125f * (float)Q[base + (size_t)qrow * D + k]);
        }

    v8f o[4] = {};
    float m[8], l[8];
#pragma unroll
    for (int r = 0; r < 8; ++r) { m[r] = -3.0e38f; l[r] = 0.0f; }

    auto stage = [&](int buf, int kvBase) {
        const __bf16* kSrc = Kg + base + (size_t)kvBase * D;
        const __bf16* vSrc = Vg + base + (size_t)kvBase * D;
#if HAVE_TDM
        if (w == 0)      tdm_load_64x64_bf16((unsigned)buf * 8192u, kSrc, D);
        else if (w == 1) tdm_load_64x64_bf16(16384u + (unsigned)buf * 8192u, vSrc, D);
#else
        __bf16* kd = KbP(buf);
        __bf16* vd = VbP(buf);
        for (int i = tid; i < 512; i += 128) {
            const int row = i >> 3, seg = i & 7;
            copy16(&kd[row * 64 + seg * 8], &kSrc[(size_t)row * D + seg * 8]);
            copy16(&vd[row * 64 + seg * 8], &vSrc[(size_t)row * D + seg * 8]);
        }
#endif
    };
    auto wait_stage = [&] { wait_tensor(); wait_async_lds(); };

    auto do_block = [&](int buf, int kvBase, bool diag) {
        const __bf16* Kb = KbP(buf);
        const __bf16* Vb = VbP(buf);

        // S = Q @ K^T : 4 key-subtiles x 2 WMMA (K-dim 64)
        v8f s[4];
#pragma unroll
        for (int nt = 0; nt < 4; ++nt) {
            v8f z = {};
            v16bf b0 = load_frag_b_nk(&Kb[(nt * 16) * 64], 64, lane);
            v16bf b1 = load_frag_b_nk(&Kb[(nt * 16) * 64 + 32], 64, lane);
            z = wmma_bf16(qF[0], b0, z);
            z = wmma_bf16(qF[1], b1, z);
            s[nt] = z;
        }

        if (diag) {
#pragma unroll
            for (int nt = 0; nt < 4; ++nt)
#pragma unroll
                for (int r = 0; r < 8; ++r) {
                    const int row = qBase + 16 * w + r + 8 * kh;
                    const int col = kvBase + nt * 16 + nl;
                    if (col > row) s[nt][r] = -3.0e38f;
                }
        }

        // online softmax (rows per accumulator VGPR, replicated over 16-lane halves)
#pragma unroll
        for (int r = 0; r < 8; ++r) {
            float loc = fmaxf(fmaxf(s[0][r], s[1][r]), fmaxf(s[2][r], s[3][r]));
            loc = half_reduce_max(loc);
            const float mn    = fmaxf(m[r], loc);
            const float alpha = __expf(m[r] - mn);
            float rs = 0.0f;
#pragma unroll
            for (int nt = 0; nt < 4; ++nt) {
                const float p = __expf(s[nt][r] - mn);
                s[nt][r] = p;
                rs += p;
            }
            rs = half_reduce_sum(rs);
            l[r] = l[r] * alpha + rs;
            m[r] = mn;
#pragma unroll
            for (int nt = 0; nt < 4; ++nt) o[nt][r] *= alpha;
        }

        // C-layout -> A-layout transpose of P via wave-private LDS
#pragma unroll
        for (int nt = 0; nt < 4; ++nt)
#pragma unroll
            for (int r = 0; r < 8; ++r)
                Pw[(r + 8 * kh) * 64 + nt * 16 + nl] = (__bf16)s[nt][r];

        // O += P @ V
        v16bf pF[2];
#pragma unroll
        for (int f = 0; f < 2; ++f)
            pF[f] = load_frag_a(&Pw[f * 32], 64, lane);
#pragma unroll
        for (int nt = 0; nt < 4; ++nt) {
            v16bf v0 = load_frag_b_kn(&Vb[0 * 64 + nt * 16], 64, lane);
            v16bf v1 = load_frag_b_kn(&Vb[32 * 64 + nt * 16], 64, lane);
            o[nt] = wmma_bf16(pF[0], v0, o[nt]);
            o[nt] = wmma_bf16(pF[1], v1, o[nt]);
        }
    };

    const int nkv = qb + 1;
    stage(0, 0);
    wait_stage();
    __syncthreads();

    int cur = 0;
    for (int kv = 0; kv < nkv; ++kv) {
        if (kv + 1 < nkv) stage(cur ^ 1, (kv + 1) * 64);
        if (kv == nkv - 1) do_block(cur, kv * 64, true);
        else               do_block(cur, kv * 64, false);
        wait_stage();
        __syncthreads();
        cur ^= 1;
    }

    // normalize, write ctx as [b, n, h*HD+hd]  (== transpose(0,2,1,3).reshape)
#pragma unroll
    for (int r = 0; r < 8; ++r) {
        const float inv = 1.0f / l[r];
        const int row = qBase + 16 * w + r + 8 * kh;
#pragma unroll
        for (int nt = 0; nt < 4; ++nt)
            ctx[base + (size_t)row * D + nt * 16 + nl] = (__bf16)(o[nt][r] * inv);
    }
}

// ---------------------------------------------------------------------------
// Host launcher
// ---------------------------------------------------------------------------
extern "C" void kernel_launch(void* const* d_in, const int* in_sizes, int n_in,
                              void* d_out, int out_size, void* d_ws, size_t ws_size,
                              hipStream_t stream) {
    constexpr int B = 4, S = 2048, D = 1024, H = 16;
    constexpr int M = B * S;  // 8192

    const float* x  = (const float*)d_in[0];
    const float* Wq = (const float*)d_in[1];
    const float* Wk = (const float*)d_in[2];
    const float* Wv = (const float*)d_in[3];
    const float* Wo = (const float*)d_in[4];
    const float* bo = (const float*)d_in[5];

    char* ws = (char*)d_ws;
    size_t off = 0;
    auto alloc = [&](size_t bytes) -> void* {
        void* p = ws + off;
        off += (bytes + 255) & ~(size_t)255;
        return p;
    };
    __bf16* xb   = (__bf16*)alloc((size_t)M * D * 2);
    __bf16* wqb  = (__bf16*)alloc((size_t)D * D * 2);
    __bf16* wkb  = (__bf16*)alloc((size_t)D * D * 2);
    __bf16* wvb  = (__bf16*)alloc((size_t)D * D * 2);
    __bf16* wob  = (__bf16*)alloc((size_t)D * D * 2);
    __bf16* Qb   = (__bf16*)alloc((size_t)M * D * 2);
    __bf16* Kbuf = (__bf16*)alloc((size_t)M * D * 2);
    __bf16* Vbuf = (__bf16*)alloc((size_t)M * D * 2);
    __bf16* ctxb = (__bf16*)alloc((size_t)M * D * 2);

    cast_f32_bf16<<<2048, 256, 0, stream>>>(x,  xb,  M * D);
    cast_f32_bf16<<<1024, 256, 0, stream>>>(Wq, wqb, D * D);
    cast_f32_bf16<<<1024, 256, 0, stream>>>(Wk, wkb, D * D);
    cast_f32_bf16<<<1024, 256, 0, stream>>>(Wv, wvb, D * D);
    cast_f32_bf16<<<1024, 256, 0, stream>>>(Wo, wob, D * D);

    dim3 gg(D / 128, M / 128);  // (8, 64)
    gemm_xwt<true><<<gg, 256, 0, stream>>>(xb, wqb, Qb,   nullptr, M, D, D);
    gemm_xwt<true><<<gg, 256, 0, stream>>>(xb, wkb, Kbuf, nullptr, M, D, D);
    gemm_xwt<true><<<gg, 256, 0, stream>>>(xb, wvb, Vbuf, nullptr, M, D, D);

    flash_attn<<<dim3(S / 64, B * H), 128, 40960, stream>>>(Qb, Kbuf, Vbuf, ctxb);

    gemm_xwt<false><<<gg, 256, 0, stream>>>(ctxb, wob, d_out, bo, M, D, D);
}
